// GradientSafeVectorizedPartitioner_51651276702445
// MI455X (gfx1250) — matrix operand order repaired and checked by reading, not compiled, not run on hardware
//
#include <hip/hip_runtime.h>
#include <hip/hip_bf16.h>

// ---------------------------------------------------------------------------
// GradientSafeVectorizedPartitioner forward for gfx1250 (MI455X, wave32/WMMA)
//
// Decomposition:
//   K1: convert x -> bf16                (grid-stride)
//   K2: convert sel_w1[:, :128] -> bf16  (x-part of the selection MLP weight)
//   K3: XW = x @ W1x^T via v_wmma_f32_16x16x32_bf16  (done ONCE; reused by all
//       16 sequential cluster rounds). B matrix staged into LDS per block via
//       the Tensor Data Mover (TENSOR_LOAD_TO_LDS + s_wait_tensorcnt); A tile
//       staged manually. All fragments then come from LDS (ds_load_b128).
//   K4: one workgroup per batch element runs the 16 serial rounds entirely in
//       LDS: logits, gumbel seed, 2-hop BFS, size MLP, top-k picks, GRU rerun.
// ---------------------------------------------------------------------------

#define BATCH  32
#define NNODE  512
#define DFEAT  128
#define HDIM   128
#define NCLUS  16
#define SZMAX  8
#define NEGINF (-1.0e9f)
#define FEPS   (1e-8f)

typedef __attribute__((ext_vector_type(16))) __bf16 v16bf;
typedef __attribute__((ext_vector_type(8)))  __bf16 v8bf;
typedef __attribute__((ext_vector_type(8)))  float  v8f;
typedef __attribute__((ext_vector_type(4)))  unsigned int u32x4;
typedef __attribute__((ext_vector_type(4)))  int i32x4;
typedef __attribute__((ext_vector_type(8)))  int i32x8;

// ------------------------------ K1 / K2: f32 -> bf16 -----------------------

__global__ void cvt_x_kernel(const float* __restrict__ src,
                             unsigned short* __restrict__ dst, int n) {
  int i = blockIdx.x * blockDim.x + threadIdx.x;
  int stride = gridDim.x * blockDim.x;
  for (; i < n; i += stride) {
    __bf16 v = (__bf16)src[i];
    dst[i] = __builtin_bit_cast(unsigned short, v);
  }
}

// sel_w1 is [H, D+H] row-major; extract the x-part [H, D] as bf16.
__global__ void cvt_w1x_kernel(const float* __restrict__ w1,
                               unsigned short* __restrict__ dst) {
  int i = blockIdx.x * blockDim.x + threadIdx.x;   // 0 .. H*D-1
  if (i >= HDIM * DFEAT) return;
  int h = i >> 7, k = i & 127;
  __bf16 v = (__bf16)w1[h * (DFEAT + HDIM) + k];
  dst[i] = __builtin_bit_cast(unsigned short, v);
}

// ------------------------------ K3: WMMA GEMM ------------------------------
// XW[m, h] = sum_k x[m,k] * w1x[h,k];  M = B*N = 16384, N = H = 128, K = 128.
// One block = 256 threads = 8 wave32s. Block owns one 16-row M tile (LDS,
// shared across waves); wave w owns N-tile w (16 cols). K loop: 4 x 32.
// The shared 32 KB B matrix is DMA'd into LDS once per block by the TDM:
// tile == tensor and stride == dim0, so the transfer is an exact contiguous
// row-major copy (see CDNA5 ISA D# Group 0/1 bitfields).

#define LDA 136  // padded LDS row stride (bf16 elems) to dodge bank conflicts

__global__ void __launch_bounds__(256)
xw_wmma_kernel(const unsigned short* __restrict__ xb_u,
               const unsigned short* __restrict__ w1b_u,
               float* __restrict__ xw) {
  const __bf16* xb = reinterpret_cast<const __bf16*>(xb_u);

  __shared__ __bf16 ldsA[16 * LDA];
  __shared__ __bf16 ldsB[HDIM * DFEAT];   // 128 x 128 bf16 = 32 KiB

  const int tid   = threadIdx.x;
  const int mbase = blockIdx.x * 16;

  // --- TDM: DMA the full B matrix (w1x bf16, row-major) into LDS -----------
  if (tid == 0) {
    unsigned long long ga = (unsigned long long)(size_t)w1b_u;
    unsigned lds_off = (unsigned)(size_t)(&ldsB[0]);   // LDS aperture: low 32b
    u32x4 g0;
    g0.x = 1u;                                     // D#.count = 1 (valid)
    g0.y = lds_off;                                // D#.lds_addr [63:32]
    g0.z = (unsigned)(ga & 0xffffffffu);           // D#.global_addr [95:64]
    g0.w = (unsigned)((ga >> 32) & 0x01ffffffu)    // D#.global_addr [120:96]
         | (2u << 30);                             // D#.type = 2 ("image")
    i32x8 g1;
    g1[0] = (int)(1u << 16);     // data_size = 1 (2-byte elems); wg_mask = 0
    g1[1] = (int)(128u << 16);   // tensor_dim0 = 128         [79:48]
    g1[2] = (int)(128u << 16);   // tensor_dim1 = 128         [111:80]
    g1[3] = (int)(128u << 16);   // tile_dim0   = 128         [127:112]
    g1[4] = 128;                 // tile_dim1   = 128         [143:128]
    g1[5] = 128;                 // tensor_dim0_stride = 128  [207:160]
    g1[6] = 0;
    g1[7] = 0;
    i32x4 z4 = {0, 0, 0, 0};
#if defined(__clang_major__) && (__clang_major__ >= 23)
    i32x8 z8 = {0, 0, 0, 0, 0, 0, 0, 0};
    __builtin_amdgcn_tensor_load_to_lds(g0, g1, z4, z4, z8, 0);
#else
    __builtin_amdgcn_tensor_load_to_lds(g0, g1, z4, z4, 0);
#endif
    __builtin_amdgcn_s_wait_tensorcnt(0);
  }

  // Stage A tile (16 x 128 bf16): each thread copies 8 contiguous bf16.
  {
    int e = tid * 8;
    int r = e >> 7, c = e & 127;
    *(v8bf*)&ldsA[r * LDA + c] = *(const v8bf*)&xb[(size_t)(mbase + r) * DFEAT + c];
  }
  __syncthreads();

  const int wave = tid >> 5;
  const int lane = tid & 31;
  const int r    = lane & 15;   // A row / B column within tile
  const int half = lane >> 4;   // K-half select (this code's contiguous k order)
  const int ncol = wave * 16 + r;

  v8f acc = {};
#pragma unroll
  for (int kb = 0; kb < DFEAT; kb += 32) {
    // A fragment (16x32 bf16): lane(half=0) K={kb..kb+15},
    //                          lane(half=1) K={kb+16..kb+31}
    v16bf a;
    ((v8bf*)&a)[0] = *(const v8bf*)&ldsA[r * LDA + kb + 16 * half];
    ((v8bf*)&a)[1] = *(const v8bf*)&ldsA[r * LDA + kb + 16 * half + 8];
    // B fragment (32x16): lane owns column ncol, K chunk 16*half..+15,
    // contiguous in row-major ldsB[ncol, :].
    v16bf b = *(const v16bf*)&ldsB[ncol * DFEAT + kb + 16 * half];
    acc = __builtin_amdgcn_wmma_f32_16x16x32_bf16(
        /*neg_a=*/false, a, /*neg_b=*/false, b,
        /*c_mod=*/(short)0, acc, /*reuse_a=*/false, /*reuse_b=*/false);
  }

  // C/D layout: VGPR v holds M = v + 8*half (lanes 0-15 / 16-31), N = lane&15.
  const int nb = wave * 16;
#pragma unroll
  for (int v = 0; v < 8; ++v) {
    int row = mbase + v + 8 * half;
    xw[(size_t)row * HDIM + nb + (lane & 15)] = acc[v];
  }
}

// ------------------------------ K4: sequential rounds ----------------------

__device__ __forceinline__ float gumbelf(unsigned sid, unsigned bb, unsigned nn) {
  unsigned x = sid * 0x9E3779B1u ^ (bb * 0x85EBCA77u + 0x165667B1u)
             ^ (nn * 0xC2B2AE3Du + 0x27220A95u);
  x ^= x >> 16; x *= 0x7FEB352Du; x ^= x >> 15; x *= 0x846CA68Bu; x ^= x >> 16;
  float u = (float)(x >> 8) * (1.0f / 16777216.0f);
  return -__logf(-__logf(u + FEPS) + FEPS);
}

__device__ __forceinline__ float sigmoidf_(float x) {
  return 1.0f / (1.0f + __expf(-x));
}

__device__ __forceinline__ float block_fsum(volatile float* rv, float v, int tid) {
  __syncthreads();
  rv[tid] = v; __syncthreads();
  for (int s = 64; s > 0; s >>= 1) {
    if (tid < s) rv[tid] += rv[tid + s];
    __syncthreads();
  }
  float r = rv[0]; __syncthreads();
  return r;
}

__device__ __forceinline__ int block_isum(volatile int* ri, int v, int tid) {
  __syncthreads();
  ri[tid] = v; __syncthreads();
  for (int s = 64; s > 0; s >>= 1) {
    if (tid < s) ri[tid] += ri[tid + s];
    __syncthreads();
  }
  int r = ri[0]; __syncthreads();
  return r;
}

__device__ __forceinline__ int block_argmax(volatile float* rv, volatile int* ri,
                                            float v, int idx, int tid) {
  __syncthreads();
  rv[tid] = v; ri[tid] = idx; __syncthreads();
  for (int s = 64; s > 0; s >>= 1) {
    if (tid < s) {
      float ov = rv[tid + s]; int oi = ri[tid + s];
      if (ov > rv[tid] || (ov == rv[tid] && oi < ri[tid])) { rv[tid] = ov; ri[tid] = oi; }
    }
    __syncthreads();
  }
  int r = ri[0]; __syncthreads();
  return r;
}

__global__ void __launch_bounds__(128)
partition_kernel(const float* __restrict__ x, const float* __restrict__ adj,
                 const unsigned char* __restrict__ mask,
                 const float* __restrict__ sel_w1, const float* __restrict__ sel_b1,
                 const float* __restrict__ sel_w2, const float* __restrict__ sel_b2,
                 const float* __restrict__ size_w1, const float* __restrict__ size_b1,
                 const float* __restrict__ size_w2, const float* __restrict__ size_b2,
                 const float* __restrict__ gru_wih, const float* __restrict__ gru_whh,
                 const float* __restrict__ gru_bih, const float* __restrict__ gru_bhh,
                 const float* __restrict__ init_w, const float* __restrict__ init_b,
                 const float* __restrict__ xw, float* __restrict__ out) {
  __shared__ float s_ctx[HDIM], s_ctxw[HDIM], s_sf[HDIM], s_xmean[HDIM];
  __shared__ float s_sh[HDIM], s_h[HDIM];
  __shared__ float s_logits[NNODE];
  __shared__ float s_hist[NCLUS][HDIM];
  __shared__ float s_rv[128];
  __shared__ int   s_ri[128];
  __shared__ unsigned char s_avail[NNODE], s_assigned[NNODE], s_cand[NNODE];
  __shared__ unsigned char s_rem[NNODE], s_mem[NNODE];
  __shared__ int s_list[NNODE];
  __shared__ int s_cnt;
  __shared__ int s_assignedTotal;

  const int tid = threadIdx.x;     // 0..127, also the H-dim this thread owns
  const int b   = blockIdx.x;
  const unsigned char* mb = mask + (size_t)b * NNODE;
  const float* adjb = adj + (size_t)b * NNODE * NNODE;

  for (int n = tid; n < NNODE; n += 128) s_assigned[n] = 0;
  if (tid == 0) s_assignedTotal = 0;

  // total = sum(mask)
  float tloc = 0.f;
  for (int n = tid; n < NNODE; n += 128) tloc += mb[n] ? 1.f : 0.f;
  const float totalf = block_fsum(s_rv, tloc, tid);

  // x_mean (thread owns feature dim tid) and initial ctx = init_w @ x_mean + b
  {
    float s = 0.f;
    for (int n = 0; n < NNODE; ++n)
      s += mb[n] ? x[((size_t)b * NNODE + n) * DFEAT + tid] : 0.f;
    s_xmean[tid] = s / (totalf + FEPS);
  }
  __syncthreads();
  {
    float a = init_b[tid];
    const float* wr = init_w + (size_t)tid * DFEAT;
    for (int j = 0; j < DFEAT; ++j) a += wr[j] * s_xmean[j];
    s_ctx[tid] = a;
  }
  __syncthreads();

  for (int c = 0; c < NCLUS; ++c) {
    const float pct = (float)s_assignedTotal / (totalf + FEPS);
    const bool active = (pct < 0.95f) && (totalf > 0.f);

    for (int n = tid; n < NNODE; n += 128)
      s_avail[n] = (mb[n] && !s_assigned[n]) ? 1 : 0;

    // ctx part of selection MLP pre-activation: W1c @ ctx + b1
    {
      float a = sel_b1[tid];
      const float* wr = sel_w1 + (size_t)tid * (DFEAT + HDIM) + DFEAT;
      for (int j = 0; j < HDIM; ++j) a += wr[j] * s_ctx[j];
      s_ctxw[tid] = a;
    }
    __syncthreads();

    // logits[n] = w2 . relu(XW[n] + ctxw) + b2   (XW precomputed by WMMA)
    for (int n = tid; n < NNODE; n += 128) {
      const float* xr = xw + ((size_t)b * NNODE + n) * HDIM;
      float acc = sel_b2[0];
      for (int h = 0; h < HDIM; ++h)
        acc += sel_w2[h] * fmaxf(xr[h] + s_ctxw[h], 0.f);
      s_logits[n] = acc;
    }
    __syncthreads();

    // Seed: gumbel-hard argmax over available nodes
    float bv = -3.0e9f; int bi = 0;
    for (int n = tid; n < NNODE; n += 128) {
      float ml = s_avail[n] ? s_logits[n] : NEGINF;
      float v = ml + gumbelf(c * 16 + 0, b, n);
      if (v > bv) { bv = v; bi = n; }
    }
    const int seed = block_argmax(s_rv, s_ri, bv, bi, tid);
    s_sf[tid] = x[((size_t)b * NNODE + seed) * DFEAT + tid];

    // 2-hop BFS from seed, restricted to avail
    for (int n = tid; n < NNODE; n += 128) s_cand[n] = 0;
    if (tid == 0) s_cnt = 0;
    __syncthreads();
    for (int m = tid; m < NNODE; m += 128) {
      if (m != seed && s_avail[m] && adjb[(size_t)seed * NNODE + m] > 0.f) {
        s_cand[m] = 1;
        int p = atomicAdd(&s_cnt, 1);
        s_list[p] = m;
      }
    }
    __syncthreads();
    const int cnt1 = s_cnt;
    for (int i = tid; i < cnt1; i += 128)              // warm L2/L0 for hop 2
      __builtin_prefetch(&adjb[(size_t)s_list[i] * NNODE], 0, 1);
    for (int m = tid; m < NNODE; m += 128) {
      if (!s_cand[m] && m != seed && s_avail[m]) {
        bool hit = false;
        for (int i = 0; i < cnt1 && !hit; ++i)
          hit = adjb[(size_t)s_list[i] * NNODE + m] > 0.f;
        if (hit) s_cand[m] = 1;
      }
    }
    __syncthreads();

    int cl = 0;
    for (int n = tid; n < NNODE; n += 128) cl += (int)s_cand[n];
    int candc = block_isum(s_ri, cl, tid) + 1;          // + seed itself
    int maxp = candc < SZMAX ? candc : SZMAX;

    // size MLP on [seed_feat, ctx, max_possible]
    {
      float a = size_b1[tid];
      const float* wr = size_w1 + (size_t)tid * (DFEAT + HDIM + 1);
      for (int j = 0; j < DFEAT; ++j) a += wr[j] * s_sf[j];
      for (int j = 0; j < HDIM; ++j)  a += wr[DFEAT + j] * s_ctx[j];
      a += wr[DFEAT + HDIM] * (float)maxp;
      s_sh[tid] = fmaxf(a, 0.f);
    }
    __syncthreads();
    float sv = -3.0e9f; int si = 0;
    if (tid < SZMAX) {
      float a = size_b2[tid];
      const float* wr = size_w2 + (size_t)tid * HDIM;
      for (int h = 0; h < HDIM; ++h) a += wr[h] * s_sh[h];
      int valid = maxp < 1 ? 1 : (maxp > SZMAX ? SZMAX : maxp);
      float ml = (tid < valid) ? a : NEGINF;
      sv = ml + gumbelf(c * 16 + 1, b, tid);
      si = tid;
    }
    const int szidx = block_argmax(s_rv, s_ri, sv, si, tid);
    const int kextra = szidx;                           // size = szidx+1, extras = size-1

    // Iterative gumbel top-k over candidates (excluding seed)
    for (int n = tid; n < NNODE; n += 128) {
      s_mem[n] = (n == seed) ? 1 : 0;
      s_rem[n] = s_cand[n];
    }
    __syncthreads();
    for (int s = 0; s < SZMAX - 1; ++s) {
      int rl = 0;
      for (int n = tid; n < NNODE; n += 128) rl += (int)s_rem[n];
      int remc = block_isum(s_ri, rl, tid);
      if (s < kextra && remc > 0) {
        float bv2 = -3.0e9f; int bi2 = 0;
        for (int n = tid; n < NNODE; n += 128) {
          float ml = s_rem[n] ? s_logits[n] : NEGINF;
          float v = ml + gumbelf(c * 16 + 2 + s, b, n);
          if (v > bv2) { bv2 = v; bi2 = n; }
        }
        int p = block_argmax(s_rv, s_ri, bv2, bi2, tid);
        if (tid == 0) { s_mem[p] = 1; s_rem[p] = 0; }
        __syncthreads();
      }
    }
    __syncthreads();

    // Emit assignment column c; update assigned set and count
    int addl = 0;
    for (int n = tid; n < NNODE; n += 128) {
      bool memb = active && s_mem[n];
      out[((size_t)b * NNODE + n) * NCLUS + c] = memb ? 1.0f : 0.0f;
      if (memb) { s_assigned[n] = 1; addl++; }
    }
    int addc = block_isum(s_ri, addl, tid);
    if (tid == 0) { s_assignedTotal += addc; s_cnt = 0; }
    __syncthreads();
    for (int n = tid; n < NNODE; n += 128)
      if (s_mem[n]) { int p = atomicAdd(&s_cnt, 1); s_list[p] = n; }
    __syncthreads();
    const int mcnt = s_cnt;

    // cluster embedding = active ? mean(x over members) : 0
    {
      float e = 0.f;
      for (int i = 0; i < mcnt; ++i)
        e += x[((size_t)b * NNODE + s_list[i]) * DFEAT + tid];
      s_hist[c][tid] = active ? e / fmaxf((float)mcnt, 1.f) : 0.f;
    }
    __syncthreads();

    // GRU rerun over hist[0..c] starting from current ctx_hidden
    s_h[tid] = s_ctx[tid];
    __syncthreads();
    for (int t = 0; t <= c; ++t) {
      const float* xt = s_hist[t];
      float gi0 = gru_bih[tid], gi1 = gru_bih[HDIM + tid], gi2 = gru_bih[2 * HDIM + tid];
      float gh0 = gru_bhh[tid], gh1 = gru_bhh[HDIM + tid], gh2 = gru_bhh[2 * HDIM + tid];
      const float* wi0 = gru_wih + (size_t)tid * DFEAT;
      const float* wi1 = gru_wih + (size_t)(HDIM + tid) * DFEAT;
      const float* wi2 = gru_wih + (size_t)(2 * HDIM + tid) * DFEAT;
      const float* wh0 = gru_whh + (size_t)tid * HDIM;
      const float* wh1 = gru_whh + (size_t)(HDIM + tid) * HDIM;
      const float* wh2 = gru_whh + (size_t)(2 * HDIM + tid) * HDIM;
      for (int j = 0; j < DFEAT; ++j) {
        float xj = xt[j];
        gi0 += wi0[j] * xj; gi1 += wi1[j] * xj; gi2 += wi2[j] * xj;
      }
      for (int j = 0; j < HDIM; ++j) {
        float hj = s_h[j];
        gh0 += wh0[j] * hj; gh1 += wh1[j] * hj; gh2 += wh2[j] * hj;
      }
      float rg = sigmoidf_(gi0 + gh0);
      float zg = sigmoidf_(gi1 + gh1);
      float ng = tanhf(gi2 + rg * gh2);
      float hn = (1.f - zg) * ng + zg * s_h[tid];
      __syncthreads();
      s_h[tid] = hn;
      __syncthreads();
    }
    s_ctx[tid] = s_h[tid];
    __syncthreads();
  }
}

// ------------------------------ launch -------------------------------------

extern "C" void kernel_launch(void* const* d_in, const int* in_sizes, int n_in,
                              void* d_out, int out_size, void* d_ws, size_t ws_size,
                              hipStream_t stream) {
  const float* x        = (const float*)d_in[0];
  const float* adj      = (const float*)d_in[1];
  const unsigned char* mask = (const unsigned char*)d_in[2];
  const float* sel_w1   = (const float*)d_in[3];
  const float* sel_b1   = (const float*)d_in[4];
  const float* sel_w2   = (const float*)d_in[5];
  const float* sel_b2   = (const float*)d_in[6];
  const float* size_w1  = (const float*)d_in[7];
  const float* size_b1  = (const float*)d_in[8];
  const float* size_w2  = (const float*)d_in[9];
  const float* size_b2  = (const float*)d_in[10];
  const float* gru_wih  = (const float*)d_in[11];
  const float* gru_whh  = (const float*)d_in[12];
  const float* gru_bih  = (const float*)d_in[13];
  const float* gru_bhh  = (const float*)d_in[14];
  const float* init_w   = (const float*)d_in[15];
  const float* init_b   = (const float*)d_in[16];
  float* out = (float*)d_out;

  // Workspace: [0, 4MiB) x_bf16 | [4MiB, 4MiB+64KiB) w1x_bf16 | then XW f32.
  char* ws = (char*)d_ws;
  unsigned short* xb  = (unsigned short*)ws;
  unsigned short* w1b = (unsigned short*)(ws + (size_t)(4u << 20));
  float* xw = (float*)(ws + (size_t)(4u << 20) + (size_t)(64u << 10));

  const int nx = BATCH * NNODE * DFEAT;
  cvt_x_kernel<<<512, 256, 0, stream>>>(x, xb, nx);
  cvt_w1x_kernel<<<(HDIM * DFEAT + 255) / 256, 256, 0, stream>>>(sel_w1, w1b);
  xw_wmma_kernel<<<BATCH * NNODE / 16, 256, 0, stream>>>(xb, w1b, xw);
  partition_kernel<<<BATCH, 128, 0, stream>>>(
      x, adj, mask, sel_w1, sel_b1, sel_w2, sel_b2,
      size_w1, size_b1, size_w2, size_b2,
      gru_wih, gru_whh, gru_bih, gru_bhh, init_w, init_b, xw, out);
}
